// ModuleGARO_67181878444763
// MI455X (gfx1250) — matrix operand order repaired and checked by reading, not compiled, not run on hardware
//
#include <hip/hip_runtime.h>
#include <hip/hip_bf16.h>
#include <math.h>

// ---------------- problem constants ----------------
#define BB   8
#define TT   64
#define SS   64
#define NN   256
#define BT   512           // BB*TT
#define KBIG 65536         // NN*NN
#define NSPLIT 8
#define KSPLIT (KBIG / NSPLIT)   // 8192

typedef __attribute__((ext_vector_type(16))) _Float16 v16h;
typedef __attribute__((ext_vector_type(8)))  _Float16 v8h;
typedef __attribute__((ext_vector_type(8)))  float    v8f;

union F16x16 { v16h v; v8h h[2]; };

// builtin signature wants pointers to a 2x int GCC vector (64 bits)
typedef int v2i __attribute__((vector_size(2 * sizeof(int))));
typedef __attribute__((address_space(1))) v2i as1_v2i;
typedef __attribute__((address_space(3))) v2i as3_v2i;

// ---- async global->LDS copy (CDNA5), with safe fallback ----
__device__ __forceinline__ void async_copy_b64(const _Float16* gp, _Float16* lp) {
#if __has_builtin(__builtin_amdgcn_global_load_async_to_lds_b64)
    __builtin_amdgcn_global_load_async_to_lds_b64(
        (as1_v2i*)(void*)gp, (as3_v2i*)(void*)lp, 0, 0);
#else
    *(unsigned long long*)lp = *(const unsigned long long*)gp;
#endif
}
__device__ __forceinline__ void wait_async0() {
#if __has_builtin(__builtin_amdgcn_global_load_async_to_lds_b64)
#if __has_builtin(__builtin_amdgcn_s_wait_asynccnt)
    __builtin_amdgcn_s_wait_asynccnt(0);
#else
    asm volatile("s_wait_asynccnt 0" ::: "memory");
#endif
#endif
}

// =====================================================================
// K0: convert + TRANSPOSE m1W (65536x256 f32) -> w1t (256x65536 f16)
// Transposed weights make the big-GEMM B tiles K-contiguous.
// grid = (KBIG/64, NN/64), 256 threads, 64x64 tile via LDS
// =====================================================================
__global__ void k_cvt_w1t(const float* __restrict__ w, _Float16* __restrict__ wt) {
    __shared__ _Float16 tl[64 * 65];
    int kb = blockIdx.x * 64, nb = blockIdx.y * 64, t = threadIdx.x;
#pragma unroll
    for (int j = 0; j < 16; ++j) {
        int i = t + j * 256;
        int k = i >> 6, n = i & 63;
        tl[k * 65 + n] = (_Float16)w[(size_t)(kb + k) * NN + nb + n];
    }
    __syncthreads();
#pragma unroll
    for (int j = 0; j < 16; ++j) {
        int i = t + j * 256;
        int n = i >> 6, k = i & 63;
        wt[(size_t)(nb + n) * KBIG + kb + k] = tl[k * 65 + n];
    }
}

// =====================================================================
// K1: per (b,t): center columns over s, scale by 1/||xm||  -> y f16
//     (the 1/(s-1) and diag-normalization of corrcoef cancel)
// grid = BT blocks, 256 threads (one per column n)
// =====================================================================
__global__ void k_norm(const float* __restrict__ x, _Float16* __restrict__ y) {
    int bt = blockIdx.x, n = threadIdx.x;
    const float* p = x + (size_t)bt * SS * NN + n;
    float s = 0.f;
    for (int t = 0; t < SS; ++t) s += p[t * NN];
    float mu = s * (1.0f / SS);
    float q = 0.f;
    for (int t = 0; t < SS; ++t) { float d = p[t * NN] - mu; q += d * d; }
    float inv = rsqrtf(q);
    _Float16* yp = y + (size_t)bt * SS * NN + n;
    for (int t = 0; t < SS; ++t) yp[t * NN] = (_Float16)((p[t * NN] - mu) * inv);
}

// =====================================================================
// K2: fc = clip(y^T y, -1, 1) per bt, WMMA f16->f32.
// y staged in LDS TRANSPOSED [n][s] (stride 72 halfs = 36 dwords:
// conflict-free, 16B aligned) so every fragment is 2x ds_load_b128.
// grid = (32, BT); 8 waves/block, one 16x16 tile per wave.
// =====================================================================
#define YT 72
__global__ void k_corr(const _Float16* __restrict__ y, _Float16* __restrict__ fc) {
    __shared__ _Float16 ys[NN * YT];
    int bt = blockIdx.y, tg = blockIdx.x, t = threadIdx.x;
    const _Float16* yb = y + (size_t)bt * SS * NN;
    for (int i = t; i < SS * NN; i += 256)
        ys[(i & 255) * YT + (i >> 8)] = yb[i];   // transpose into LDS
    __syncthreads();

    int wave = t >> 5, lane = t & 31;
    int id = tg * 8 + wave;
    int i0 = (id >> 4) * 16, j0 = (id & 15) * 16;
    int m = lane & 15, half = lane >> 4;   // m = A-row (M) and B/C-col (N)

    v8f c = {0.f, 0.f, 0.f, 0.f, 0.f, 0.f, 0.f, 0.f};
#pragma unroll
    for (int kk = 0; kk < SS; kk += 32) {
        F16x16 a, b;
        // A (16x32 MxK): lane holds A[m][kk+half*8+w] and A[m][kk+16+half*8+w]
        a.h[0] = *(const v8h*)&ys[(i0 + m) * YT + kk + half * 8];
        a.h[1] = *(const v8h*)&ys[(i0 + m) * YT + kk + 16 + half * 8];
        // B (32x16 KxN): lane holds B[kk+half*8+w][n] and B[kk+16+half*8+w][n]
        b.h[0] = *(const v8h*)&ys[(j0 + m) * YT + kk + half * 8];
        b.h[1] = *(const v8h*)&ys[(j0 + m) * YT + kk + 16 + half * 8];
        c = __builtin_amdgcn_wmma_f32_16x16x32_f16(false, a.v, false, b.v,
                                                   (short)0, c, false, false);
    }
    _Float16* out = fc + (size_t)bt * NN * NN;
#pragma unroll
    for (int r = 0; r < 8; ++r) {
        float v = c[r];
        v = fminf(fmaxf(v, -1.0f), 1.0f);
        out[(size_t)(i0 + r + half * 8) * NN + j0 + m] = (_Float16)v;
    }
}

// =====================================================================
// K3: big GEMM  fc(512x65536 f16) @ m1W -> via TRANSPOSED w1t(256x65536).
// split-K=8; grid=(16 coltiles, 4 rowgroups, 8 splits), 8 waves share the
// B tile (16 rows x 64 K, LDS [n][k] stride 72). B staged with per-lane
// contiguous 8-byte async global->LDS copies. A = 2x global_load_b128.
// K-step 64 = 2 WMMAs per barrier pair. Partials -> ws (det. reduce).
// =====================================================================
#define BSTR 72
__global__ void k_gemm1(const _Float16* __restrict__ fc, const _Float16* __restrict__ wt,
                        float* __restrict__ part) {
    __shared__ _Float16 bs[16 * BSTR];   // [n][k], k = 0..63
    int jt = blockIdx.x, rg = blockIdx.y, sp = blockIdx.z, t = threadIdx.x;
    int wave = t >> 5, lane = t & 31;
    int m = lane & 15, half = lane >> 4;
    int row0 = (rg * 8 + wave) * 16;
    int col0 = jt * 16;
    int k0   = sp * KSPLIT;

    // staging mapping: thread t copies 4 halfs (8B) of row n=t>>4 at k=(t&15)*4
    int ln = t >> 4, lk = (t & 15) * 4;
    const _Float16* gsrc = wt + (size_t)(col0 + ln) * KBIG + k0 + lk;
    _Float16* ldst = &bs[ln * BSTR + lk];

    v8f c = {0.f, 0.f, 0.f, 0.f, 0.f, 0.f, 0.f, 0.f};
    // A row base for this lane: fc[row0+m][k0 + half*8 + ...], K-contiguous
    const _Float16* pa = fc + (size_t)(row0 + m) * KBIG + k0 + half * 8;

    for (int kk = 0; kk < KSPLIT; kk += 64) {
        async_copy_b64(gsrc + kk, ldst);     // B tile -> LDS (async on CDNA5)
        wait_async0();
        __syncthreads();

        F16x16 a0, a1, b0, b1;
        a0.h[0] = *(const v8h*)(pa + kk);
        a0.h[1] = *(const v8h*)(pa + kk + 16);
        a1.h[0] = *(const v8h*)(pa + kk + 32);
        a1.h[1] = *(const v8h*)(pa + kk + 48);
        b0.h[0] = *(const v8h*)&bs[m * BSTR + half * 8];
        b0.h[1] = *(const v8h*)&bs[m * BSTR + 16 + half * 8];
        b1.h[0] = *(const v8h*)&bs[m * BSTR + 32 + half * 8];
        b1.h[1] = *(const v8h*)&bs[m * BSTR + 48 + half * 8];
        __builtin_prefetch(pa + kk + 128, 0, 0);   // next A chunk

        c = __builtin_amdgcn_wmma_f32_16x16x32_f16(false, a0.v, false, b0.v,
                                                   (short)0, c, false, false);
        c = __builtin_amdgcn_wmma_f32_16x16x32_f16(false, a1.v, false, b1.v,
                                                   (short)0, c, false, false);
        __syncthreads();
    }

    float* o = part + (size_t)sp * BT * NN;
#pragma unroll
    for (int r = 0; r < 8; ++r)
        o[(size_t)(row0 + r + half * 8) * NN + col0 + m] = c[r];
}

// =====================================================================
// K4: reduce split-K partials + bias -> H1 (512x256 f32)
// =====================================================================
__global__ void k_red(const float* __restrict__ part, const float* __restrict__ bias,
                      float* __restrict__ H) {
    int i = blockIdx.x * 256 + threadIdx.x;
    float s = bias[i & 255];
    for (int sp = 0; sp < NSPLIT; ++sp) s += part[(size_t)sp * BT * NN + i];
    H[i] = s;
}

// BatchNorm batch statistics over 512 rows (biased variance); 1 block
__global__ void k_stats(const float* __restrict__ H, float* __restrict__ mean,
                        float* __restrict__ var) {
    int c = threadIdx.x;
    float s = 0.f, q = 0.f;
    for (int r = 0; r < BT; ++r) { float v = H[r * NN + c]; s += v; q += v * v; }
    float mu = s * (1.0f / BT);
    mean[c] = mu;
    var[c]  = q * (1.0f / BT) - mu * mu;
}

// BN(train) + ReLU, elementwise. grid = BT
__global__ void k_bnrelu(const float* __restrict__ H, const float* __restrict__ mean,
                         const float* __restrict__ var, const float* __restrict__ g,
                         const float* __restrict__ be, float* __restrict__ A) {
    int i = blockIdx.x * 256 + threadIdx.x;
    int c = i & 255;
    float v = (H[i] - mean[c]) * rsqrtf(var[c] + 1e-5f) * g[c] + be[c];
    A[i] = fmaxf(v, 0.f);
}

// small GEMM: A(rows x 256) @ W(256x256) + bias.  grid = rows
__global__ void k_gemm_small(const float* __restrict__ A, const float* __restrict__ W,
                             const float* __restrict__ bias, float* __restrict__ H) {
    int row = blockIdx.x, col = threadIdx.x;
    float acc = bias[col];
    for (int k = 0; k < 256; ++k) acc += A[row * NN + k] * W[k * NN + col];
    H[row * NN + col] = acc;
}

// GEMM with concatenated input [A(512x256) | broadcast e_s] @ p1W(512x256)+p1b
__global__ void k_gemm_cat(const float* __restrict__ A, const float* __restrict__ es,
                           const float* __restrict__ W, const float* __restrict__ bias,
                           float* __restrict__ H) {
    int row = blockIdx.x, col = threadIdx.x, b = row >> 6;
    float acc = bias[col];
    for (int k = 0; k < 256; ++k) acc += A[row * NN + k] * W[k * NN + col];
    for (int k = 0; k < 256; ++k) acc += es[b * NN + k] * W[(256 + k) * NN + col];
    H[row * NN + col] = acc;
}

// mean over T per batch. grid = BB
__global__ void k_meanT(const float* __restrict__ Hh, float* __restrict__ Hm) {
    int b = blockIdx.x, c = threadIdx.x;
    float s = 0.f;
    for (int t = 0; t < TT; ++t) s += Hh[(size_t)(b * TT + t) * NN + c];
    Hm[b * NN + c] = s * (1.0f / TT);
}

// att[bt] = sigmoid( dot(xq[b], xk[bt]) / 16 ).  grid = 2
__global__ void k_att(const float* __restrict__ xq, const float* __restrict__ xk,
                      float* __restrict__ att) {
    int bt = blockIdx.x * 256 + threadIdx.x;
    int b = bt >> 6;
    float dot = 0.f;
    for (int c = 0; c < 256; ++c) dot += xq[b * NN + c] * xk[bt * NN + c];
    att[bt] = 1.0f / (1.0f + expf(-dot * 0.0625f));
}

// out = Hh * att broadcast. grid = BT
__global__ void k_out(const float* __restrict__ Hh, const float* __restrict__ att,
                      float* __restrict__ out) {
    int i = blockIdx.x * 256 + threadIdx.x;
    out[i] = Hh[i] * att[i >> 8];
}

// =====================================================================
extern "C" void kernel_launch(void* const* d_in, const int* in_sizes, int n_in,
                              void* d_out, int out_size, void* d_ws, size_t ws_size,
                              hipStream_t stream) {
    const float* x     = (const float*)d_in[0];
    const float* e_s   = (const float*)d_in[1];
    const float* qW    = (const float*)d_in[2];
    const float* qb    = (const float*)d_in[3];
    const float* kW    = (const float*)d_in[4];
    const float* kb    = (const float*)d_in[5];
    const float* m1W   = (const float*)d_in[6];
    const float* m1b   = (const float*)d_in[7];
    const float* m1g   = (const float*)d_in[8];
    const float* m1be  = (const float*)d_in[9];
    const float* m2W   = (const float*)d_in[10];
    const float* m2b   = (const float*)d_in[11];
    const float* m2g   = (const float*)d_in[12];
    const float* m2be  = (const float*)d_in[13];
    const float* p1W   = (const float*)d_in[14];
    const float* p1b   = (const float*)d_in[15];
    const float* p1g   = (const float*)d_in[16];
    const float* p1be  = (const float*)d_in[17];
    const float* p2W   = (const float*)d_in[18];
    const float* p2b   = (const float*)d_in[19];
    const float* p2g   = (const float*)d_in[20];
    const float* p2be  = (const float*)d_in[21];
    float* out = (float*)d_out;

    // ---- workspace carve-up (256B aligned) ----
    char* ws = (char*)d_ws;
    size_t off = 0;
    auto carve = [&](size_t bytes) -> char* {
        char* p = ws + off;
        off = (off + bytes + 255) & ~(size_t)255;
        return p;
    };
    _Float16* y16  = (_Float16*)carve((size_t)BT * SS * NN * 2);      // 16.8 MB
    _Float16* fc16 = (_Float16*)carve((size_t)BT * KBIG * 2);         // 67.1 MB
    _Float16* w1t  = (_Float16*)carve((size_t)KBIG * NN * 2);         // 33.6 MB (transposed)
    float* part = (float*)carve((size_t)NSPLIT * BT * NN * 4);        // 4.2 MB
    float* H1   = (float*)carve((size_t)BT * NN * 4);
    float* H2   = (float*)carve((size_t)BT * NN * 4);
    float* H3   = (float*)carve((size_t)BT * NN * 4);
    float* H4   = (float*)carve((size_t)BT * NN * 4);
    float* A1   = (float*)carve((size_t)BT * NN * 4);
    float* A2   = (float*)carve((size_t)BT * NN * 4);
    float* A3   = (float*)carve((size_t)BT * NN * 4);
    float* Hh   = (float*)carve((size_t)BT * NN * 4);
    float* st   = (float*)carve((size_t)8 * NN * 4);
    float* Hm   = (float*)carve((size_t)BB * NN * 4);
    float* xq   = (float*)carve((size_t)BB * NN * 4);
    float* xk   = (float*)carve((size_t)BT * NN * 4);
    float* att  = (float*)carve((size_t)BT * 4);
    float *mean1 = st,            *var1 = st + NN;
    float *mean2 = st + 2 * NN,   *var2 = st + 3 * NN;
    float *mean3 = st + 4 * NN,   *var3 = st + 5 * NN;
    float *mean4 = st + 6 * NN,   *var4 = st + 7 * NN;

    // 0) weight convert + transpose
    k_cvt_w1t<<<dim3(KBIG / 64, NN / 64), 256, 0, stream>>>(m1W, w1t);
    // 1) normalize samples
    k_norm<<<BT, 256, 0, stream>>>(x, y16);
    // 2) correlation GEMM (WMMA)
    k_corr<<<dim3(32, BT), 256, 0, stream>>>(y16, fc16);
    // 3) big GEMM (WMMA, split-K, async B staging)
    k_gemm1<<<dim3(16, 4, NSPLIT), 256, 0, stream>>>(fc16, w1t, part);
    k_red<<<BT, 256, 0, stream>>>(part, m1b, H1);
    // 4) BN/ReLU chain + small GEMMs
    k_stats<<<1, 256, 0, stream>>>(H1, mean1, var1);
    k_bnrelu<<<BT, 256, 0, stream>>>(H1, mean1, var1, m1g, m1be, A1);
    k_gemm_small<<<BT, 256, 0, stream>>>(A1, m2W, m2b, H2);
    k_stats<<<1, 256, 0, stream>>>(H2, mean2, var2);
    k_bnrelu<<<BT, 256, 0, stream>>>(H2, mean2, var2, m2g, m2be, A2);   // fc_seq
    k_gemm_cat<<<BT, 256, 0, stream>>>(A2, e_s, p1W, p1b, H3);
    k_stats<<<1, 256, 0, stream>>>(H3, mean3, var3);
    k_bnrelu<<<BT, 256, 0, stream>>>(H3, mean3, var3, p1g, p1be, A3);
    k_gemm_small<<<BT, 256, 0, stream>>>(A3, p2W, p2b, H4);
    k_stats<<<1, 256, 0, stream>>>(H4, mean4, var4);
    k_bnrelu<<<BT, 256, 0, stream>>>(H4, mean4, var4, p2g, p2be, Hh);
    // 5) sigmoid query-key attention over time
    k_meanT<<<BB, 256, 0, stream>>>(Hh, Hm);
    k_gemm_small<<<BB, 256, 0, stream>>>(Hm, qW, qb, xq);
    k_gemm_small<<<BT, 256, 0, stream>>>(Hh, kW, kb, xk);
    k_att<<<2, 256, 0, stream>>>(xq, xk, att);
    // 6) output
    k_out<<<BT, 256, 0, stream>>>(Hh, att, out);
}
